// SGNS_20555713479270
// MI455X (gfx1250) — matrix operand order repaired and checked by reading, not compiled, not run on hardware
//
#include <hip/hip_runtime.h>
#include <cmath>

#define DDIM 128
#define WAVES 8

typedef __attribute__((ext_vector_type(16))) _Float16 v16h;
typedef __attribute__((ext_vector_type(8)))  _Float16 v8h;
typedef __attribute__((ext_vector_type(4)))  _Float16 v4h;
typedef __attribute__((ext_vector_type(8)))  float    v8f;

__device__ __forceinline__ float log_sigmoid(float x) {
    // stable: min(x,0) - log1p(exp(-|x|))
    return fminf(x, 0.0f) - log1pf(__expf(-fabsf(x)));
}

__device__ __forceinline__ void cvt_store4(_Float16* dst, float4 g) {
    v4h h;
    h[0] = (_Float16)g.x; h[1] = (_Float16)g.y;
    h[2] = (_Float16)g.z; h[3] = (_Float16)g.w;
    *(v4h*)dst = h;            // 8-byte LDS store of packed f16
}

// One wave per (center, context) pair. 6 dot products per pair via a WMMA
// accumulator chain: A rows 0..5 = [ov, nv0..nv4] (rows 6..15 are duplicate
// rows, never read back from D), B = center vector broadcast to all 16
// columns, C/D accumulate in f32. Embeddings are converted f32->f16 once at
// staging; fragments are then pure ds_load_b128 of packed f16.
__global__ __launch_bounds__(WAVES * 32)
void sgns_pair_kernel(const int* __restrict__ iword,
                      const int* __restrict__ owords,
                      const int* __restrict__ nwords,
                      const float* __restrict__ emb_i,
                      const float* __restrict__ emb_o,
                      float* __restrict__ partial,
                      int NP, int W, int NNEG)
{
    __shared__ _Float16 s_ctx[WAVES][6][DDIM];   // f16 context rows
    __shared__ _Float16 s_iv[WAVES][DDIM];       // f16 center row
    __shared__ float    s_sum[WAVES];

    const int wave = threadIdx.x >> 5;
    const int lane = threadIdx.x & 31;
    const int p    = blockIdx.x * WAVES + wave;   // pair index (wave-uniform)
    const bool active = (p < NP);

    if (active) {
        // Coalesced 512B row gathers; convert f32->f16 in-flight (RTNE).
        const int center = iword[p / W];
        cvt_store4(&s_iv[wave][lane * 4],
                   ((const float4*)(emb_i + (size_t)center * DDIM))[lane]);

        const int pos = owords[p];
        cvt_store4(&s_ctx[wave][0][lane * 4],
                   ((const float4*)(emb_o + (size_t)pos * DDIM))[lane]);

        #pragma unroll
        for (int j = 0; j < 5; ++j) {
            const int neg = nwords[(size_t)p * NNEG + j];
            cvt_store4(&s_ctx[wave][1 + j][lane * 4],
                       ((const float4*)(emb_o + (size_t)neg * DDIM))[lane]);
        }
    }
    __syncthreads();

    if (active) {
        const int m    = lane & 15;   // A-matrix row this lane feeds
        const int half = lane >> 4;   // lane half selects K sub-range
        // Rows 6..15 never read back from D: feed duplicate rows (finite).
        const int row  = (m < 6) ? m : ((m < 12) ? m - 6 : m - 12);
        const _Float16* crow = &s_ctx[wave][row][0];
        const _Float16* iv   = &s_iv[wave][0];

        v8f acc = {};
        #pragma unroll
        for (int kc = 0; kc < 4; ++kc) {            // K = 128 in 4 x 32 chunks
            const int kb = kc * 32;
            // A fragment: two contiguous 8xf16 runs (16B each, 16B aligned).
            v8h alo = *(const v8h*)(crow + kb + half * 8);        // K kb+8h..
            v8h ahi = *(const v8h*)(crow + kb + 16 + half * 8);   // K kb+16+8h..
            v16h a  = __builtin_shufflevector(alo, ahi,
                        0, 1, 2, 3, 4, 5, 6, 7, 8, 9, 10, 11, 12, 13, 14, 15);
            // B fragment (broadcast center): two 8xf16 runs matching A's K order, 16B aligned.
            v8h blo = *(const v8h*)(iv + kb + half * 8);
            v8h bhi = *(const v8h*)(iv + kb + 16 + half * 8);
            v16h bm = __builtin_shufflevector(blo, bhi,
                        0, 1, 2, 3, 4, 5, 6, 7, 8, 9, 10, 11, 12, 13, 14, 15);

            acc = __builtin_amdgcn_wmma_f32_16x16x32_f16(
                false, a, false, bm, (short)0, acc, false, false);
        }

        // D layout: VGPR r, lanes 0-15 -> row r. Lane 0 holds all 6 dots.
        if (lane == 0) {
            float loss = log_sigmoid(acc[0]);                 // positive pair
            #pragma unroll
            for (int j = 1; j <= 5; ++j)
                loss += log_sigmoid(-acc[j]);                 // negatives
            s_sum[wave] = loss;
        }
    } else if (lane == 0) {
        s_sum[wave] = 0.0f;
    }
    __syncthreads();

    if (threadIdx.x == 0) {
        float s = 0.0f;
        #pragma unroll
        for (int w = 0; w < WAVES; ++w) s += s_sum[w];
        partial[blockIdx.x] = s;   // one deterministic partial per block
    }
}

// Deterministic parallel reduction: fixed per-thread ranges + fixed-pairing
// LDS tree (bitwise reproducible across runs).
__global__ __launch_bounds__(256)
void sgns_final_reduce(const float* __restrict__ partial, int n,
                       float inv_b, float* __restrict__ out)
{
    __shared__ float s[256];
    const int t = threadIdx.x;
    const int per = (n + 255) / 256;
    float v = 0.0f;
    for (int i = 0; i < per; ++i) {
        const int idx = t * per + i;
        if (idx < n) v += partial[idx];
    }
    s[t] = v;
    __syncthreads();
    #pragma unroll
    for (int off = 128; off > 0; off >>= 1) {
        if (t < off) s[t] += s[t + off];
        __syncthreads();
    }
    if (t == 0) out[0] = -s[0] * inv_b;
}

extern "C" void kernel_launch(void* const* d_in, const int* in_sizes, int n_in,
                              void* d_out, int out_size, void* d_ws, size_t ws_size,
                              hipStream_t stream) {
    const int*   iword  = (const int*)d_in[0];
    const int*   owords = (const int*)d_in[1];
    const int*   nwords = (const int*)d_in[2];
    const float* emb_i  = (const float*)d_in[3];
    const float* emb_o  = (const float*)d_in[4];
    float*       out    = (float*)d_out;

    const int Bsz  = in_sizes[0];             // 4096
    const int NP   = in_sizes[1];             // B*W = 16384
    const int W    = NP / Bsz;                // 4
    const int NNEG = in_sizes[2] / NP;        // 5

    float* partial = (float*)d_ws;
    const int nblocks = (NP + WAVES - 1) / WAVES;

    sgns_pair_kernel<<<nblocks, WAVES * 32, 0, stream>>>(
        iword, owords, nwords, emb_i, emb_o, partial, NP, W, NNEG);
    sgns_final_reduce<<<1, 256, 0, stream>>>(partial, nblocks,
                                             1.0f / (float)Bsz, out);
}
